// SparseVGG16Regression_27530740367954
// MI455X (gfx1250) — compile-verified, matrix operand order checked by reading, not run on hardware
//
#include <hip/hip_runtime.h>

typedef __attribute__((ext_vector_type(16))) __bf16 v16bf;
typedef __attribute__((ext_vector_type(8)))  __bf16 v8bf;
typedef __attribute__((ext_vector_type(8)))  float  v8f;

#define CONV_BLOCK 256
#define MT 128            // pixel tile
#define NT 64             // out-channel tile
#define KT 64             // k per stage (2 WMMA depths)
#define LDAe 72           // padded LDS row stride (bf16 elems): 144B, 16B-aligned
#define LDBe 72

// ---------------- CDNA5 async global->LDS copy (ASYNCcnt-tracked) -----------------
__device__ __forceinline__ void async_b128(const __bf16* g, __bf16* l) {
  unsigned int       loff = (unsigned int)(uintptr_t)l;        // LDS byte offset (aperture low 32)
  unsigned long long ga   = (unsigned long long)(uintptr_t)g;
  asm volatile("global_load_async_to_lds_b128 %0, %1, off"
               :: "v"(loff), "v"(ga)
               : "memory");
}
__device__ __forceinline__ void wait_async0() {
  asm volatile("s_wait_asynccnt 0x0" ::: "memory");
}

// ---------------- fragment gather from LDS ----------------------------------------
// 16-bit A layout (16x32): lanes 0-15 = rows, K 0..7 (v0-3) / 16..23 (v4-7);
// lanes 16-31 same rows, K 8..15 / 24..31.  B from [N][K] tile, mirrored pattern.
__device__ __forceinline__ v16bf frag_from_lds(const __bf16* rowbase, int koff) {
  v8bf lo = *(const v8bf*)(rowbase + koff);
  v8bf hi = *(const v8bf*)(rowbase + koff + 16);
  v16bf r;
#pragma unroll
  for (int i = 0; i < 8; ++i) { r[i] = lo[i]; r[i + 8] = hi[i]; }
  return r;
}

// ---------------- weight prep: OIHW fp32 -> [Cout][(kh*3+kw)*Cin + ci] bf16 -------
__global__ __launch_bounds__(256) void prep_w_kernel(const float* __restrict__ w,
                                                     __bf16* __restrict__ wb,
                                                     int Cin, int Cout) {
  int idx = blockIdx.x * 256 + threadIdx.x;
  int total = Cout * Cin * 9;
  if (idx >= total) return;
  int r  = idx % 9;
  int t  = idx / 9;
  int ci = t % Cin;
  int co = t / Cin;
  wb[(size_t)co * (9 * Cin) + r * Cin + ci] = (__bf16)w[idx];
}

// ---------------- layer 1 (Cin=1): direct conv + mask generation ------------------
__global__ __launch_bounds__(256) void conv1_kernel(const float* __restrict__ x,
                                                    const float* __restrict__ w,
                                                    const float* __restrict__ b,
                                                    __bf16* __restrict__ out,
                                                    float* __restrict__ mask,
                                                    int H, int W) {
  __shared__ float sw[64 * 9];
  __shared__ float sb[64];
  for (int i = threadIdx.x; i < 64 * 9; i += 256) sw[i] = w[i];
  if (threadIdx.x < 64) sb[threadIdx.x] = b[threadIdx.x];
  __syncthreads();

  const int p  = blockIdx.x * 256 + threadIdx.x;   // grid exact
  const int xx = p % W;
  const int yy = (p / W) % H;
  const int bb = p / (W * H);

  float xin[9];
#pragma unroll
  for (int r = 0; r < 9; ++r) {
    const int sy = yy + r / 3 - 1;
    const int sx = xx + r % 3 - 1;
    xin[r] = (sy >= 0 && sy < H && sx >= 0 && sx < W)
                 ? x[((size_t)bb * H + sy) * W + sx] : 0.0f;
  }
  const float m = (xin[4] != 0.0f) ? 1.0f : 0.0f;
  mask[p] = m;
  __bf16* orow = out + (size_t)p * 64;
#pragma unroll 4
  for (int co = 0; co < 64; ++co) {
    float acc = sb[co];
#pragma unroll
    for (int r = 0; r < 9; ++r) acc += xin[r] * sw[co * 9 + r];
    acc = (m != 0.0f) ? fmaxf(acc, 0.0f) : 0.0f;
    orow[co] = (__bf16)acc;
  }
}

// ---------------- implicit-GEMM conv via WMMA bf16, async double-buffered ---------
// out[p][co] = relu( mask[p] * ( sum_k A[p][k] * Wt[co][k] + bias[co] ) )
// A: NHWC bf16, k = (kh*3+kw)*CIN + ci.  64 | CIN, so one stage = one tap slice.
template <int CIN>
__global__ __launch_bounds__(CONV_BLOCK) void conv_wmma_kernel(
    const __bf16* __restrict__ in,   // [B,H,W,CIN]
    const __bf16* __restrict__ wb,   // [Cout][9*CIN]
    const float*  __restrict__ bias, // [Cout]
    const float*  __restrict__ mask, // [B,H,W]
    __bf16* __restrict__ out,        // [B,H,W,Cout]
    int H, int W, int shw, int Cout)
{
  constexpr int K      = 9 * CIN;
  constexpr int NSTEPS = K / KT;                               // 9,18,36,72
  constexpr int SH     = (CIN == 64) ? 0 : (CIN == 128) ? 1 : (CIN == 256) ? 2 : 3;

  __shared__ __bf16 sA[2][MT * LDAe];   // 2 x 18KB
  __shared__ __bf16 sB[2][NT * LDBe];   // 2 x 9KB

  const int m0 = blockIdx.x * MT;       // pixel base
  const int n0 = blockIdx.y * NT;       // out-channel base

  const int tid  = threadIdx.x;
  const int lane = tid & 31;
  const int wave = tid >> 5;            // 8 waves: 4 in M x 2 in N
  const int wm = wave & 3;
  const int wn = wave >> 2;

  // A staging: 2 threads per pixel, 64B (32 bf16) each
  const int pl   = tid >> 1;
  const int half = tid & 1;
  const int p    = m0 + pl;
  const int px   = p & (W - 1);
  const int py   = (p >> shw) & (H - 1);
  const int pb   = p >> (2 * shw);

  // B staging: 4 threads per out-channel, 32B (16 bf16) each
  const int nl = tid >> 2;
  const int q  = tid & 3;
  const __bf16* wrow = wb + (size_t)(n0 + nl) * K + q * 16;

  auto stage = [&](int s, int b) {
    const int r   = s >> SH;                         // tap 0..8
    const int ci0 = (s & ((CIN >> 6) - 1)) << 6;     // channel base within tap
    const int dy  = r / 3 - 1;
    const int dx  = r - (r / 3) * 3 - 1;
    const int yy  = py + dy;
    const int xx  = px + dx;
    __bf16* la = &sA[b][pl * LDAe + half * 32];
    if (yy >= 0 && yy < H && xx >= 0 && xx < W) {
      const __bf16* ga = in + (((size_t)(pb * H + yy) * W + xx) * CIN + ci0 + half * 32);
      async_b128(ga,      la);
      async_b128(ga + 8,  la + 8);
      async_b128(ga + 16, la + 16);
      async_b128(ga + 24, la + 24);
    } else {
      uint4 z = {0, 0, 0, 0};
      *(uint4*)(la)      = z;
      *(uint4*)(la + 8)  = z;
      *(uint4*)(la + 16) = z;
      *(uint4*)(la + 24) = z;
    }
    const __bf16* gb = wrow + s * KT;
    __bf16* lb = &sB[b][nl * LDBe + q * 16];
    async_b128(gb,     lb);
    async_b128(gb + 8, lb + 8);
  };

  v8f acc[2][2] = {};
  const int rc    = lane & 15;
  const int kharf = (lane >> 4) << 3;

  stage(0, 0);
  int buf = 0;
#pragma unroll 1
  for (int s = 0; s < NSTEPS; ++s) {
    wait_async0();          // my stage(s) landed
    __syncthreads();        // everyone's stage(s) landed; prior reads of buf^1 done
    if (s + 1 < NSTEPS) stage(s + 1, buf ^ 1);   // stream next tile during compute

    const __bf16* Abase = sA[buf];
    const __bf16* Bbase = sB[buf];
#pragma unroll
    for (int c = 0; c < 2; ++c) {
      const int koff = c * 32 + kharf;
      v16bf a0 = frag_from_lds(Abase + (wm * 32 + rc) * LDAe, koff);
      v16bf a1 = frag_from_lds(Abase + (wm * 32 + 16 + rc) * LDAe, koff);
      v16bf b0 = frag_from_lds(Bbase + (wn * 32 + rc) * LDBe, koff);
      v16bf b1 = frag_from_lds(Bbase + (wn * 32 + 16 + rc) * LDBe, koff);
      acc[0][0] = __builtin_amdgcn_wmma_f32_16x16x32_bf16(false, a0, false, b0, (short)0, acc[0][0], false, false);
      acc[0][1] = __builtin_amdgcn_wmma_f32_16x16x32_bf16(false, a0, false, b1, (short)0, acc[0][1], false, false);
      acc[1][0] = __builtin_amdgcn_wmma_f32_16x16x32_bf16(false, a1, false, b0, (short)0, acc[1][0], false, false);
      acc[1][1] = __builtin_amdgcn_wmma_f32_16x16x32_bf16(false, a1, false, b1, (short)0, acc[1][1], false, false);
    }
    buf ^= 1;
  }

  // ---- epilogue: bias + mask + relu, store NHWC bf16 ----
  // C/D layout: vgpr v, lane l -> row = v + 8*(l>=16), col = l&15
  const int col   = lane & 15;
  const int rbase = (lane >> 4) * 8;
#pragma unroll
  for (int mi = 0; mi < 2; ++mi) {
#pragma unroll
    for (int v = 0; v < 8; ++v) {
      const int pix = m0 + wm * 32 + mi * 16 + rbase + v;
      const float mv = mask[pix];
#pragma unroll
      for (int ni = 0; ni < 2; ++ni) {
        const int ch = n0 + wn * 32 + ni * 16 + col;
        float val = acc[mi][ni][v] + bias[ch];
        val = (mv != 0.0f) ? fmaxf(val, 0.0f) : 0.0f;
        out[(size_t)pix * Cout + ch] = (__bf16)val;
      }
    }
  }
}

// ---------------- 2x2 maxpool on NHWC bf16 ----------------------------------------
__global__ __launch_bounds__(256) void pool_act_kernel(const __bf16* __restrict__ in,
                                                       __bf16* __restrict__ out,
                                                       int Ho, int Wo, int C) {
  size_t idx = (size_t)blockIdx.x * 256 + threadIdx.x;
  size_t total = (size_t)2 * Ho * Wo * C;
  if (idx >= total) return;
  int c = idx % C; size_t t = idx / C;
  int xo = t % Wo; t /= Wo;
  int yo = t % Ho;
  int b  = t / Ho;
  const int Hi = 2 * Ho, Wi = 2 * Wo;
  const __bf16* base = in + (((size_t)b * Hi + 2 * yo) * Wi + 2 * xo) * C + c;
  float v00 = (float)base[0];
  float v01 = (float)base[(size_t)C];
  float v10 = (float)base[(size_t)Wi * C];
  float v11 = (float)base[(size_t)Wi * C + C];
  out[idx] = (__bf16)fmaxf(fmaxf(v00, v01), fmaxf(v10, v11));
}

__global__ __launch_bounds__(256) void pool_mask_kernel(const float* __restrict__ in,
                                                        float* __restrict__ out,
                                                        int Ho, int Wo) {
  size_t idx = (size_t)blockIdx.x * 256 + threadIdx.x;
  size_t total = (size_t)2 * Ho * Wo;
  if (idx >= total) return;
  int xo = idx % Wo; size_t t = idx / Wo;
  int yo = t % Ho;
  int b  = t / Ho;
  const int Hi = 2 * Ho, Wi = 2 * Wo;
  const float* base = in + ((size_t)b * Hi + 2 * yo) * Wi + 2 * xo;
  out[idx] = fmaxf(fmaxf(base[0], base[1]), fmaxf(base[Wi], base[Wi + 1]));
}

// ---------------- flatten NHWC bf16 [2,16,16,512] -> NCHW-order fp32 --------------
__global__ __launch_bounds__(256) void flatten_kernel(const __bf16* __restrict__ in,
                                                      float* __restrict__ out) {
  int idx = blockIdx.x * 256 + threadIdx.x;   // grid exact
  int b = idx >> 17;
  int k = idx & 131071;
  int c = k >> 8;
  int rem = k & 255;
  int y = rem >> 4;
  int x = rem & 15;
  out[idx] = (float)in[((((size_t)b * 16 + y) * 16 + x) * 512) + c];
}

// ---------------- FC: out[b][n] = (relu)(in[b]·w[n] + bias[n]) --------------------
__global__ __launch_bounds__(256) void fc_kernel(const float* __restrict__ in,
                                                 const float* __restrict__ w,
                                                 const float* __restrict__ bias,
                                                 float* __restrict__ out,
                                                 int K, int N, int relu) {
  const int n = blockIdx.x;
  const int b = blockIdx.y;
  const float* wr = w + (size_t)n * K;
  const float* ir = in + (size_t)b * K;
  float s = 0.0f;
  const int k4 = K & ~(4 * 256 - 1);
  for (int k = threadIdx.x * 4; k < k4; k += 1024) {
    float4 a = *(const float4*)(ir + k);
    float4 c = *(const float4*)(wr + k);
    s += a.x * c.x + a.y * c.y + a.z * c.z + a.w * c.w;
  }
  for (int k = k4 + threadIdx.x; k < K; k += 256) s += ir[k] * wr[k];

  __shared__ float red[256];
  red[threadIdx.x] = s;
  __syncthreads();
  for (int o = 128; o > 0; o >>= 1) {
    if (threadIdx.x < o) red[threadIdx.x] += red[threadIdx.x + o];
    __syncthreads();
  }
  if (threadIdx.x == 0) {
    float v = red[0] + bias[n];
    out[(size_t)b * N + n] = relu ? fmaxf(v, 0.0f) : v;
  }
}

// ==================================================================================
extern "C" void kernel_launch(void* const* d_in, const int* in_sizes, int n_in,
                              void* d_out, int out_size, void* d_ws, size_t ws_size,
                              hipStream_t stream) {
  (void)in_sizes; (void)n_in; (void)out_size; (void)ws_size;
  static const int CHv[14] = {1,64,64,128,128,256,256,256,512,512,512,512,512,512};

  const float* x = (const float*)d_in[0];
  const float* fw[4] = {(const float*)d_in[27], (const float*)d_in[29],
                        (const float*)d_in[31], (const float*)d_in[33]};
  const float* fb[4] = {(const float*)d_in[28], (const float*)d_in[30],
                        (const float*)d_in[32], (const float*)d_in[34]};

  // ---- carve workspace ----
  char* ws = (char*)d_ws;
  size_t off = 0;
  auto carve = [&](size_t bytes) -> char* {
    char* pp = ws + off;
    off = (off + bytes + 255) & ~(size_t)255;
    return pp;
  };
  __bf16* wbL[14] = {};
  for (int l = 2; l <= 13; ++l)
    wbL[l] = (__bf16*)carve((size_t)CHv[l] * CHv[l - 1] * 9 * sizeof(__bf16));
  __bf16* actA = (__bf16*)carve((size_t)2 * 512 * 512 * 64 * sizeof(__bf16));
  __bf16* actB = (__bf16*)carve((size_t)2 * 512 * 512 * 64 * sizeof(__bf16));
  float*  maskA = (float*)carve((size_t)2 * 512 * 512 * sizeof(float));
  float*  maskB = (float*)carve((size_t)2 * 512 * 512 * sizeof(float));
  float*  flat  = (float*)carve((size_t)2 * 131072 * sizeof(float));
  float*  fc1o  = (float*)carve((size_t)2 * 1024 * sizeof(float));
  float*  fc2o  = (float*)carve((size_t)2 * 512 * sizeof(float));
  float*  fc3o  = (float*)carve((size_t)2 * 256 * sizeof(float));

  // ---- prep bf16 [Cout][9*Cin] weights for layers 2..13 ----
  for (int l = 2; l <= 13; ++l) {
    const float* wsrc = (const float*)d_in[2 * l - 1];
    int total = CHv[l] * CHv[l - 1] * 9;
    prep_w_kernel<<<(total + 255) / 256, 256, 0, stream>>>(wsrc, wbL[l], CHv[l - 1], CHv[l]);
  }

  // ---- layer 1 direct + mask ----
  conv1_kernel<<<(2 * 512 * 512) / 256, 256, 0, stream>>>(
      x, (const float*)d_in[1], (const float*)d_in[2], actA, maskA, 512, 512);

  // ---- layers 2..13 (WMMA implicit GEMM), pooling after 2,4,7,10,13 ----
  const __bf16* cur = actA;  __bf16* oth = actB;
  const float*  mcur = maskA; float*  moth = maskB;
  int H = 512;
  for (int l = 2; l <= 13; ++l) {
    const int Cin = CHv[l - 1], Cout = CHv[l];
    const int M = 2 * H * H;
    const float* bconv = (const float*)d_in[2 * l];
    int shw = 31 - __builtin_clz((unsigned)H);
    dim3 grid((unsigned)(M / MT), (unsigned)(Cout / NT));
    switch (Cin) {
      case 64:
        conv_wmma_kernel<64><<<grid, CONV_BLOCK, 0, stream>>>(cur, wbL[l], bconv, mcur, oth, H, H, shw, Cout);
        break;
      case 128:
        conv_wmma_kernel<128><<<grid, CONV_BLOCK, 0, stream>>>(cur, wbL[l], bconv, mcur, oth, H, H, shw, Cout);
        break;
      case 256:
        conv_wmma_kernel<256><<<grid, CONV_BLOCK, 0, stream>>>(cur, wbL[l], bconv, mcur, oth, H, H, shw, Cout);
        break;
      default:
        conv_wmma_kernel<512><<<grid, CONV_BLOCK, 0, stream>>>(cur, wbL[l], bconv, mcur, oth, H, H, shw, Cout);
        break;
    }
    { const __bf16* t = cur; cur = oth; oth = (__bf16*)t; }

    if (l == 2 || l == 4 || l == 7 || l == 10 || l == 13) {
      const int Ho = H / 2;
      size_t tot = (size_t)2 * Ho * Ho * Cout;
      pool_act_kernel<<<(unsigned)((tot + 255) / 256), 256, 0, stream>>>(cur, oth, Ho, Ho, Cout);
      { const __bf16* t = cur; cur = oth; oth = (__bf16*)t; }
      size_t mt = (size_t)2 * Ho * Ho;
      pool_mask_kernel<<<(unsigned)((mt + 255) / 256), 256, 0, stream>>>(mcur, moth, Ho, Ho);
      { const float* t = mcur; mcur = moth; moth = (float*)t; }
      H = Ho;
    }
  }

  // ---- head: flatten (NHWC -> NCHW order) + 4 FC layers ----
  flatten_kernel<<<(2 * 131072) / 256, 256, 0, stream>>>(cur, flat);
  fc_kernel<<<dim3(1024, 2), 256, 0, stream>>>(flat, fw[0], fb[0], fc1o, 131072, 1024, 1);
  fc_kernel<<<dim3(512, 2),  256, 0, stream>>>(fc1o, fw[1], fb[1], fc2o, 1024, 512, 1);
  fc_kernel<<<dim3(256, 2),  256, 0, stream>>>(fc2o, fw[2], fb[2], fc3o, 512, 256, 1);
  fc_kernel<<<dim3(2, 2),    256, 0, stream>>>(fc3o, fw[3], fb[3], (float*)d_out, 256, 2, 0);
}
